// HierDSFeedForward_71451075936553
// MI455X (gfx1250) — compile-verified, matrix operand order checked by reading, not run on hardware
//
#include <hip/hip_runtime.h>
#include <stdint.h>

// ---------------- problem constants ----------------
#define S_TOK 4096      // B*T
#define C_DIM 1024      // model dim
#define H_DIM 2048      // hidden
#define NEXP  8
#define EPG   4
#define AST   40        // LDS tile row stride in bf16 elements (80B = 20 banks)

typedef __attribute__((ext_vector_type(16))) __bf16 v16bf;
typedef __attribute__((ext_vector_type(8)))  float  v8f;
typedef int vint4 __attribute__((vector_size(16)));   // matches builtin param type

union FragBits { int4 q[2]; v16bf v; };

// ---- async global->LDS path (guarded; sync fallback keeps compile green) ----
#if defined(__AMDGCN__) && \
    __has_builtin(__builtin_amdgcn_global_load_async_to_lds_b128) && \
    __has_builtin(__builtin_amdgcn_s_wait_asynccnt)
#define USE_ASYNC_LDS 1
#define WAIT_ASYNC(n) __builtin_amdgcn_s_wait_asynccnt(n)
#else
#define USE_ASYNC_LDS 0
#define WAIT_ASYNC(n) ((void)0)
#endif

__device__ __forceinline__ void copy16(const uint16_t* __restrict__ g, uint16_t* l) {
#if USE_ASYNC_LDS
  __builtin_amdgcn_global_load_async_to_lds_b128(
      (__attribute__((address_space(1))) vint4*)g,
      (__attribute__((address_space(3))) vint4*)l, 0, 0);
#else
  *reinterpret_cast<int4*>(l) = *reinterpret_cast<const int4*>(g);
#endif
}

// fp32 -> bf16 round-to-nearest-even (bits)
__device__ __forceinline__ uint16_t f2bf(float f) {
  union { float f; uint32_t u; } c; c.f = f;
  uint32_t u = c.u;
  if ((u & 0x7F800000u) == 0x7F800000u) {                 // inf / nan
    uint16_t h = (uint16_t)(u >> 16);
    if (u & 0x007FFFFFu) h |= 0x0040;                     // keep NaN quiet
    return h;
  }
  return (uint16_t)((u + 0x7FFFu + ((u >> 16) & 1u)) >> 16);
}

__device__ __forceinline__ v8f wmma_bf16(v16bf a, v16bf b, v8f c) {
  return __builtin_amdgcn_wmma_f32_16x16x32_bf16(false, a, false, b,
                                                 (short)0, c, false, false);
}

// Fragment from an LDS tile (row-major, AST-element row stride).
// Per ISA 7.12.2 (16-bit A/B layout): lane l (l<16): row=l, K={0..7,16..23};
// lane l+16: K+=8.  Two 16B ds loads per lane.
__device__ __forceinline__ v16bf lds_frag(const uint16_t* s, int row0) {
  int lane = threadIdx.x & 31;
  int r = lane & 15, hi = lane >> 4;
  const uint16_t* p = s + (row0 + r) * AST + 8 * hi;
  FragBits f;
  f.q[0] = *reinterpret_cast<const int4*>(p);
  f.q[1] = *reinterpret_cast<const int4*>(p + 16);
  return f.v;
}

// ---------------- conversion (16B vectorized) ----------------
__global__ void cvt_bf16_kernel(const float* __restrict__ src,
                                uint16_t* __restrict__ dst, int n) {
  int i = (blockIdx.x * blockDim.x + threadIdx.x) * 8;
  int stride = gridDim.x * blockDim.x * 8;
  for (; i < n; i += stride) {
    float4 f0 = *reinterpret_cast<const float4*>(src + i);
    float4 f1 = *reinterpret_cast<const float4*>(src + i + 4);
    union { uint16_t h[8]; int4 q; } o;
    o.h[0] = f2bf(f0.x); o.h[1] = f2bf(f0.y); o.h[2] = f2bf(f0.z); o.h[3] = f2bf(f0.w);
    o.h[4] = f2bf(f1.x); o.h[5] = f2bf(f1.y); o.h[6] = f2bf(f1.z); o.h[7] = f2bf(f1.w);
    *reinterpret_cast<int4*>(dst + i) = o.q;
  }
}

// ---------------- gating: one wave per token ----------------
__global__ __launch_bounds__(256) void gate_kernel(
    const float* __restrict__ x, const float* __restrict__ wg,
    const float* __restrict__ we, const float* __restrict__ gb,
    const float* __restrict__ eb,
    int* __restrict__ topi, float* __restrict__ topv) {
  int s = blockIdx.x * 8 + (threadIdx.x >> 5);
  int lane = threadIdx.x & 31;
  if (s >= S_TOK) return;
  const float* xr = x + (size_t)s * C_DIM;
  float acc[10];
  #pragma unroll
  for (int j = 0; j < 10; ++j) acc[j] = 0.f;
  for (int i = lane; i < C_DIM; i += 32) {
    float xv = xr[i];
    acc[0] += xv * wg[i];
    acc[1] += xv * wg[C_DIM + i];
    #pragma unroll
    for (int e = 0; e < NEXP; ++e) acc[2 + e] += xv * we[e * C_DIM + i];
  }
  #pragma unroll
  for (int j = 0; j < 10; ++j)
    #pragma unroll
    for (int off = 16; off > 0; off >>= 1)
      acc[j] += __shfl_down(acc[j], off, 32);
  if (lane == 0) {
    float g0 = acc[0] + gb[0], g1 = acc[1] + gb[1];
    int gidx = (g1 > g0) ? 1 : 0;                       // argmax, first on tie
    float mg = fmaxf(g0, g1);
    float p0 = __expf(g0 - mg), p1 = __expf(g1 - mg);
    float gp = (gidx ? p1 : p0) / (p0 + p1);
    int base = gidx * EPG;
    float el[4], pe[4];
    float me = -1e30f;
    #pragma unroll
    for (int j = 0; j < 4; ++j) {
      el[j] = acc[2 + base + j] + eb[base + j];
      me = fmaxf(me, el[j]);
    }
    float es = 0.f;
    #pragma unroll
    for (int j = 0; j < 4; ++j) { pe[j] = __expf(el[j] - me); es += pe[j]; }
    #pragma unroll
    for (int j = 0; j < 4; ++j) pe[j] = pe[j] / es * gp;
    int i0 = 0;
    #pragma unroll
    for (int j = 1; j < 4; ++j) if (pe[j] > pe[i0]) i0 = j;
    int i1 = -1;
    #pragma unroll
    for (int j = 0; j < 4; ++j)
      if (j != i0 && (i1 < 0 || pe[j] > pe[i1])) i1 = j;
    topi[2 * s]     = base + i0;
    topi[2 * s + 1] = base + i1;
    topv[2 * s]     = pe[i0];
    topv[2 * s + 1] = pe[i1];
  }
}

// ---------------- deterministic ballot scatter ----------------
__global__ __launch_bounds__(256) void scatter_kernel(
    const int* __restrict__ topi, const float* __restrict__ topv,
    int* __restrict__ lidx, float* __restrict__ lw, int* __restrict__ counts) {
  int e = threadIdx.x >> 5;
  int lane = threadIdx.x & 31;
  int b0 = 0, b1 = 0;
  unsigned lt = (lane == 31) ? 0x7FFFFFFFu : ((1u << lane) - 1u);
  for (int t0 = 0; t0 < S_TOK; t0 += 32) {
    int s = t0 + lane;
    bool m0 = (topi[2 * s] == e);
    bool m1 = (topi[2 * s + 1] == e);
    unsigned msk0 = (unsigned)__ballot(m0);
    unsigned msk1 = (unsigned)__ballot(m1);
    if (m0) {
      int p = b0 + __popc(msk0 & lt);
      lidx[(e * 2) * S_TOK + p] = s;
      lw[(e * 2) * S_TOK + p] = topv[2 * s];
    }
    if (m1) {
      int p = b1 + __popc(msk1 & lt);
      lidx[(e * 2 + 1) * S_TOK + p] = s;
      lw[(e * 2 + 1) * S_TOK + p] = topv[2 * s + 1];
    }
    b0 += __popc(msk0);
    b1 += __popc(msk1);
  }
  if (lane == 0) { counts[e * 2] = b0; counts[e * 2 + 1] = b1; }
}

// ---------------- fused X @ W^T + swiglu -> bf16 h ----------------
// WG = 8 waves, tile 64 tok x 128 h-cols. LDS double-buffered:
// sA 64x32, sB 256x32 (a-rows colWG..+127, b-rows H+colWG..+127).
// All B fragments of a K-step are loaded up front (flat array, no phis) so
// the ds_load clause overlaps the WMMA burst with graduated dscnt waits.
__global__ __launch_bounds__(256) void gemm1_swiglu_kernel(
    const uint16_t* __restrict__ xb, const uint16_t* __restrict__ wb,
    uint16_t* __restrict__ hout) {
  __shared__ __attribute__((aligned(16))) uint16_t sA[2][64 * AST];
  __shared__ __attribute__((aligned(16))) uint16_t sB[2][256 * AST];
  int tid = threadIdx.x;
  int wave = tid >> 5, lane = tid & 31;
  int rowWG = blockIdx.y * 64;
  int colWG = blockIdx.x * 128;
  int crow = tid >> 2, cchk = (tid & 3) * 8;   // copy: row, 8-elem chunk

  const uint16_t* gA = xb + (size_t)(rowWG + crow) * C_DIM + cchk;
  const uint16_t* gB = wb + (size_t)(colWG + crow) * C_DIM + cchk;
  uint16_t* lA  = &sA[0][crow * AST + cchk];
  uint16_t* lB0 = &sB[0][crow * AST + cchk];

  auto copy_tile = [&](int buf, int k0) {
    uint16_t* la = lA + (size_t)buf * 64 * AST;
    uint16_t* lb = lB0 + (size_t)buf * 256 * AST;
    copy16(gA + k0, la);
    copy16(gB + k0, lb);
    copy16(gB + (size_t)64 * C_DIM + k0, lb + 64 * AST);
    copy16(gB + (size_t)H_DIM * C_DIM + k0, lb + 128 * AST);
    copy16(gB + (size_t)(H_DIM + 64) * C_DIM + k0, lb + 192 * AST);
  };

  int row0l = (wave & 3) * 16;
  int col0l = (wave >> 2) * 64;
  // B fragment order: j<4 -> a-part col block j; j>=4 -> b-part block j-4
  auto brow = [&](int j) {
    return (j < 4) ? (col0l + 16 * j) : (128 + col0l + 16 * (j - 4));
  };
  v8f zero = {0.f, 0.f, 0.f, 0.f, 0.f, 0.f, 0.f, 0.f};
  v8f acc[8];
  #pragma unroll
  for (int i = 0; i < 8; ++i) acc[i] = zero;

  copy_tile(0, 0);
  const int nsteps = C_DIM / 32;
  for (int s = 0; s < nsteps; ++s) {
    int cur = s & 1;
    if (s + 1 < nsteps) { copy_tile(cur ^ 1, (s + 1) * 32); WAIT_ASYNC(5); }
    else                { WAIT_ASYNC(0); }
    __syncthreads();
    v16bf a = lds_frag(sA[cur], row0l);
    v16bf bf[8];
    #pragma unroll
    for (int j = 0; j < 8; ++j) bf[j] = lds_frag(sB[cur], brow(j));
    #pragma unroll
    for (int j = 0; j < 8; ++j) acc[j] = wmma_bf16(a, bf[j], acc[j]);
    __syncthreads();
  }

  int n = lane & 15, hi = lane >> 4;
  #pragma unroll
  for (int i = 0; i < 4; ++i)
    #pragma unroll
    for (int v = 0; v < 8; ++v) {
      float av = acc[i][v], bv = acc[4 + i][v];
      float sv = av / (1.f + __expf(-av)) * bv;          // silu(a)*b
      int r = rowWG + row0l + v + 8 * hi;
      hout[(size_t)r * H_DIM + colWG + col0l + 16 * i + n] = f2bf(sv);
    }
}

// ---------------- shared down-proj: h @ w_out^T -> d_out ----------------
__global__ __launch_bounds__(256) void gemm2_shared_kernel(
    const uint16_t* __restrict__ h, const uint16_t* __restrict__ wob,
    float* __restrict__ out) {
  __shared__ __attribute__((aligned(16))) uint16_t sA[2][64 * AST];
  __shared__ __attribute__((aligned(16))) uint16_t sB[2][128 * AST];
  int tid = threadIdx.x;
  int wave = tid >> 5, lane = tid & 31;
  int rowWG = blockIdx.y * 64;
  int colWG = blockIdx.x * 128;
  int crow = tid >> 2, cchk = (tid & 3) * 8;

  const uint16_t* gA = h + (size_t)(rowWG + crow) * H_DIM + cchk;
  const uint16_t* gB = wob + (size_t)(colWG + crow) * H_DIM + cchk;
  uint16_t* lA = &sA[0][crow * AST + cchk];
  uint16_t* lB = &sB[0][crow * AST + cchk];

  auto copy_tile = [&](int buf, int k0) {
    uint16_t* la = lA + (size_t)buf * 64 * AST;
    uint16_t* lb = lB + (size_t)buf * 128 * AST;
    copy16(gA + k0, la);
    copy16(gB + k0, lb);
    copy16(gB + (size_t)64 * H_DIM + k0, lb + 64 * AST);
  };

  int row0l = (wave & 3) * 16;
  int col0l = (wave >> 2) * 64;
  v8f zero = {0.f, 0.f, 0.f, 0.f, 0.f, 0.f, 0.f, 0.f};
  v8f acc[4];
  #pragma unroll
  for (int i = 0; i < 4; ++i) acc[i] = zero;

  copy_tile(0, 0);
  const int nsteps = H_DIM / 32;
  for (int s = 0; s < nsteps; ++s) {
    int cur = s & 1;
    if (s + 1 < nsteps) { copy_tile(cur ^ 1, (s + 1) * 32); WAIT_ASYNC(3); }
    else                { WAIT_ASYNC(0); }
    __syncthreads();
    v16bf a = lds_frag(sA[cur], row0l);
    v16bf bf[4];
    #pragma unroll
    for (int j = 0; j < 4; ++j) bf[j] = lds_frag(sB[cur], col0l + 16 * j);
    #pragma unroll
    for (int j = 0; j < 4; ++j) acc[j] = wmma_bf16(a, bf[j], acc[j]);
    __syncthreads();
  }

  int n = lane & 15, hi = lane >> 4;
  #pragma unroll
  for (int i = 0; i < 4; ++i)
    #pragma unroll
    for (int v = 0; v < 8; ++v) {
      int r = rowWG + row0l + v + 8 * hi;
      out[(size_t)r * C_DIM + colWG + col0l + 16 * i + n] = acc[i][v];
    }
}

// ---------------- routed down-proj over gathered tokens ----------------
__global__ __launch_bounds__(256) void gemm2_routed_kernel(
    const uint16_t* __restrict__ h, const uint16_t* __restrict__ w2b,
    const int* __restrict__ lidx, const float* __restrict__ lw,
    const int* __restrict__ counts,
    float* __restrict__ r0, float* __restrict__ r1) {
  int combo = blockIdx.z;
  int e = combo >> 1, slot = combo & 1;
  int cnt = counts[combo];
  int rowWG = blockIdx.y * 64;
  if (rowWG >= cnt) return;                              // WG-uniform exit
  __shared__ __attribute__((aligned(16))) uint16_t sA[2][64 * AST];
  __shared__ __attribute__((aligned(16))) uint16_t sB[2][128 * AST];
  int tid = threadIdx.x;
  int wave = tid >> 5, lane = tid & 31;
  int colWG = blockIdx.x * 128;
  const int*   il = lidx + (size_t)combo * S_TOK;
  const float* wl = lw   + (size_t)combo * S_TOK;
  const uint16_t* wexp = w2b + (size_t)e * C_DIM * H_DIM;

  int crow = tid >> 2, cchk = (tid & 3) * 8;
  int crowg = rowWG + crow;
  int tokc = il[crowg < cnt ? crowg : (cnt - 1)];        // clamp tail rows
  const uint16_t* gA = h + (size_t)tokc * H_DIM + cchk;
  const uint16_t* gB = wexp + (size_t)(colWG + crow) * H_DIM + cchk;
  uint16_t* lA = &sA[0][crow * AST + cchk];
  uint16_t* lB = &sB[0][crow * AST + cchk];

  auto copy_tile = [&](int buf, int k0) {
    uint16_t* la = lA + (size_t)buf * 64 * AST;
    uint16_t* lb = lB + (size_t)buf * 128 * AST;
    copy16(gA + k0, la);
    copy16(gB + k0, lb);
    copy16(gB + (size_t)64 * H_DIM + k0, lb + 64 * AST);
  };

  int row0l = (wave & 3) * 16;
  int col0l = (wave >> 2) * 64;
  v8f zero = {0.f, 0.f, 0.f, 0.f, 0.f, 0.f, 0.f, 0.f};
  v8f acc[4];
  #pragma unroll
  for (int i = 0; i < 4; ++i) acc[i] = zero;

  copy_tile(0, 0);
  const int nsteps = H_DIM / 32;
  for (int s = 0; s < nsteps; ++s) {
    int cur = s & 1;
    if (s + 1 < nsteps) { copy_tile(cur ^ 1, (s + 1) * 32); WAIT_ASYNC(3); }
    else                { WAIT_ASYNC(0); }
    __syncthreads();
    v16bf a = lds_frag(sA[cur], row0l);
    v16bf bf[4];
    #pragma unroll
    for (int j = 0; j < 4; ++j) bf[j] = lds_frag(sB[cur], col0l + 16 * j);
    #pragma unroll
    for (int j = 0; j < 4; ++j) acc[j] = wmma_bf16(a, bf[j], acc[j]);
    __syncthreads();
  }

  float* dst = slot ? r1 : r0;
  int n = lane & 15, hi = lane >> 4;
  #pragma unroll
  for (int i = 0; i < 4; ++i)
    #pragma unroll
    for (int v = 0; v < 8; ++v) {
      int rr = rowWG + row0l + v + 8 * hi;
      if (rr < cnt) {
        int tk = il[rr];
        dst[(size_t)tk * C_DIM + colWG + col0l + 16 * i + n] = acc[i][v] * wl[rr];
      }
    }
}

// ---------------- final combine (fixed order -> deterministic) ----------
__global__ void add_kernel(float* __restrict__ out, const float* __restrict__ r0,
                           const float* __restrict__ r1, int n) {
  int i = (blockIdx.x * blockDim.x + threadIdx.x) * 4;
  if (i < n) {
    float4 o = *reinterpret_cast<float4*>(out + i);
    float4 a = *reinterpret_cast<const float4*>(r0 + i);
    float4 b = *reinterpret_cast<const float4*>(r1 + i);
    o.x += a.x + b.x; o.y += a.y + b.y; o.z += a.z + b.z; o.w += a.w + b.w;
    *reinterpret_cast<float4*>(out + i) = o;
  }
}

// ---------------- host orchestration ----------------
extern "C" void kernel_launch(void* const* d_in, const int* in_sizes, int n_in,
                              void* d_out, int out_size, void* d_ws, size_t ws_size,
                              hipStream_t stream) {
  (void)in_sizes; (void)n_in; (void)out_size; (void)ws_size;
  const float* x     = (const float*)d_in[0];
  const float* w_in  = (const float*)d_in[1];
  const float* w_out = (const float*)d_in[2];
  const float* w1    = (const float*)d_in[3];
  const float* w2    = (const float*)d_in[4];
  const float* wg    = (const float*)d_in[5];
  const float* we    = (const float*)d_in[6];
  const float* gb    = (const float*)d_in[7];
  const float* eb    = (const float*)d_in[8];
  float* out = (float*)d_out;

  char* ws = (char*)d_ws;
  size_t off = 0;
  auto carve = [&](size_t bytes) -> void* {
    void* p = ws + off;
    off += (bytes + 255) & ~(size_t)255;
    return p;
  };
  uint16_t* xb    = (uint16_t*)carve((size_t)S_TOK * C_DIM * 2);
  uint16_t* winb  = (uint16_t*)carve((size_t)2 * H_DIM * C_DIM * 2);
  uint16_t* w1b   = (uint16_t*)carve((size_t)2 * H_DIM * C_DIM * 2);
  uint16_t* woutb = (uint16_t*)carve((size_t)C_DIM * H_DIM * 2);
  uint16_t* w2b   = (uint16_t*)carve((size_t)NEXP * C_DIM * H_DIM * 2);
  uint16_t* h_sh  = (uint16_t*)carve((size_t)S_TOK * H_DIM * 2);
  uint16_t* h_rt  = (uint16_t*)carve((size_t)S_TOK * H_DIM * 2);
  float*    rt0   = (float*)carve((size_t)S_TOK * C_DIM * 4);
  float*    rt1   = (float*)carve((size_t)S_TOK * C_DIM * 4);
  int*      topi  = (int*)carve((size_t)S_TOK * 2 * 4);
  float*    topv  = (float*)carve((size_t)S_TOK * 2 * 4);
  int*      lidx  = (int*)carve((size_t)16 * S_TOK * 4);
  float*    lw    = (float*)carve((size_t)16 * S_TOK * 4);
  int*      cnts  = (int*)carve(64 * 4);

  auto cvt = [&](const float* s, uint16_t* d, int n) {
    int blocks = (n / 8 + 255) / 256;
    if (blocks > 4096) blocks = 4096;
    cvt_bf16_kernel<<<blocks, 256, 0, stream>>>(s, d, n);
  };
  cvt(x,     xb,    S_TOK * C_DIM);
  cvt(w_in,  winb,  2 * H_DIM * C_DIM);
  cvt(w1,    w1b,   2 * H_DIM * C_DIM);
  cvt(w_out, woutb, C_DIM * H_DIM);
  cvt(w2,    w2b,   NEXP * C_DIM * H_DIM);

  gate_kernel<<<S_TOK / 8, 256, 0, stream>>>(x, wg, we, gb, eb, topi, topv);
  scatter_kernel<<<1, 256, 0, stream>>>(topi, topv, lidx, lw, cnts);

  gemm1_swiglu_kernel<<<dim3(H_DIM / 128, S_TOK / 64), 256, 0, stream>>>(xb, winb, h_sh);
  gemm1_swiglu_kernel<<<dim3(H_DIM / 128, S_TOK / 64), 256, 0, stream>>>(xb, w1b, h_rt);

  gemm2_shared_kernel<<<dim3(C_DIM / 128, S_TOK / 64), 256, 0, stream>>>(h_sh, woutb, out);
  gemm2_routed_kernel<<<dim3(C_DIM / 128, S_TOK / 64, 16), 256, 0, stream>>>(
      h_rt, w2b, lidx, lw, cnts, rt0, rt1);

  add_kernel<<<(S_TOK * C_DIM / 4 + 255) / 256, 256, 0, stream>>>(out, rt0, rt1, S_TOK * C_DIM);
}
